// RGN_88965952569972
// MI455X (gfx1250) — matrix operand order repaired and checked by reading, not compiled, not run on hardware
//
#include <hip/hip_runtime.h>

// ---------------------------------------------------------------------------
// RGN (biLSTM x2 + angle head + NeRF) for MI455X / gfx1250
// wave32 + v_wmma_f32_16x16x32_bf16 + global_load_async_to_lds_b128 pipelines
// ---------------------------------------------------------------------------

#define DEVI __device__ __forceinline__

typedef __bf16 bf16;
typedef __bf16 v16bf __attribute__((ext_vector_type(16)));
typedef float  v8f   __attribute__((ext_vector_type(8)));
typedef unsigned uv4 __attribute__((ext_vector_type(4)));
typedef unsigned uv2 __attribute__((ext_vector_type(2)));

union BfPack { uv4 u[2]; v16bf v; };

static constexpr int LSEQ = 512;
static constexpr int BN   = 32;
static constexpr int HID  = 800;
static constexpr int G4   = 3200;     // 4*H
static constexpr int KP0  = 64;       // padded D_IN (42 -> 64)
static constexpr int M1   = LSEQ * BN;  // 16384 rows for pre-GEMM1
static constexpr int NWG  = 25;       // workgroups per direction (25*32 = 800 units)
static constexpr int SW_STRIDE = 808; // LDS row stride (bf16) for Whh slice, padded

// pre-GEMM1 async pipeline geometry
static constexpr int PG_COLS   = 64;
static constexpr int PG_CK     = 160;                    // K per chunk (5 k-blocks)
static constexpr int PG_NCHUNK = 10;                     // 1600 / 160
static constexpr int PG_CHUNKB = PG_COLS * PG_CK * 2;    // 20480 bytes / buffer

DEVI unsigned bfbits(float x) {
  unsigned u = __builtin_bit_cast(unsigned, x);
  return (u + 0x7FFFu + ((u >> 16) & 1u)) >> 16;   // round-to-nearest-even
}
DEVI void bf8_to_f(uv4 u, float* o) {
#pragma unroll
  for (int i = 0; i < 4; ++i) {
    o[2*i]   = __builtin_bit_cast(float, u[i] << 16);
    o[2*i+1] = __builtin_bit_cast(float, u[i] & 0xFFFF0000u);
  }
}
DEVI uv4 f_to_bf8(const float* f) {
  uv4 o;
#pragma unroll
  for (int i = 0; i < 4; ++i)
    o[i] = bfbits(f[2*i]) | (bfbits(f[2*i+1]) << 16);
  return o;
}
DEVI float sigm(float x) { return 1.0f / (1.0f + expf(-x)); }

DEVI BfPack load_pack(const bf16* p0, const bf16* p1) {
  BfPack r;
  r.u[0] = *(const uv4*)p0;
  r.u[1] = *(const uv4*)p1;
  return r;
}

// single-lane async global->LDS 16-byte move (GVS addressing: saddr + 32b voffset)
DEVI void async_g2l_b128(unsigned lds_byte, unsigned gofs_byte, const void* sbase) {
  asm volatile("global_load_async_to_lds_b128 %0, %1, %2"
               :: "v"(lds_byte), "v"(gofs_byte), "s"(sbase) : "memory");
}
DEVI void wait_async0() {
  asm volatile("s_wait_asynccnt 0x0" ::: "memory");
}

// ---------------------------------------------------------------------------
// prep kernels
// ---------------------------------------------------------------------------
__global__ void k_cvt_pad(const float* __restrict__ src, bf16* __restrict__ dst,
                          int rows, int scols, int dcols) {
  int i = blockIdx.x * blockDim.x + threadIdx.x;
  int total = rows * dcols;
  if (i >= total) return;
  int r = i / dcols, c = i - r * dcols;
  float v = (c < scols) ? src[(size_t)r * scols + c] : 0.0f;
  ((unsigned short*)dst)[i] = (unsigned short)bfbits(v);
}

__global__ void k_bias_sum(const float* __restrict__ a, const float* __restrict__ b,
                           float* __restrict__ dst, int n) {
  int i = blockIdx.x * blockDim.x + threadIdx.x;
  if (i < n) dst[i] = a[i] + b[i];
}

__global__ void k_init_bar(int* bar) {
  if (threadIdx.x < 8) bar[threadIdx.x] = 0;
}

// ---------------------------------------------------------------------------
// grid (per-direction) spin barrier
// ---------------------------------------------------------------------------
DEVI void grid_barrier(int* bar, int nwg) {
  __threadfence();
  __syncthreads();
  if (threadIdx.x == 0) {
    int gen = __hip_atomic_load(&bar[1], __ATOMIC_ACQUIRE, __HIP_MEMORY_SCOPE_AGENT);
    int old = __hip_atomic_fetch_add(&bar[0], 1, __ATOMIC_ACQ_REL, __HIP_MEMORY_SCOPE_AGENT);
    if (old == nwg - 1) {
      __hip_atomic_store(&bar[0], 0, __ATOMIC_RELAXED, __HIP_MEMORY_SCOPE_AGENT);
      __hip_atomic_fetch_add(&bar[1], 1, __ATOMIC_ACQ_REL, __HIP_MEMORY_SCOPE_AGENT);
    } else {
      while (__hip_atomic_load(&bar[1], __ATOMIC_ACQUIRE, __HIP_MEMORY_SCOPE_AGENT) == gen) {
        __builtin_amdgcn_s_sleep(1);
      }
    }
  }
  __syncthreads();
}

// ---------------------------------------------------------------------------
// Persistent biLSTM layer kernel.  grid = 2 dirs * 25 WGs, block = 256 (8 waves).
// Each WG owns hidden units [32*wg, 32*wg+32) -> all 4 gate column blocks.
// Whh slice (128 cols x 800 K bf16 ~ 205 KB) is DMA'd into LDS once via
// global_load_async_to_lds_b128 and reused for all 512 timesteps.
// Wave w: gate = w>>1, mtile (batch 16-rows) = w&1, two 16x16 output tiles.
// Inner K-loop is 1-deep software pipelined: loads for kb+1 issue before the
// two WMMAs of kb.
// ---------------------------------------------------------------------------
template <int LAYER>
__global__ void __launch_bounds__(256)
k_lstm(const bf16* __restrict__ whh,    // [dir][3200*800]  (layer base)
       const bf16* __restrict__ wih0,   // [dir][3200*64]   (layer0 only)
       const bf16* __restrict__ xbf,    // [512*32*64]      (layer0 only)
       const bf16* __restrict__ gpreT,  // [dir][3200][16384] (layer1 only)
       const float* __restrict__ bias,  // [dir][3200] combined (layer0 adds)
       bf16* __restrict__ hstate,       // [dir][buf2][32][800]
       bf16* __restrict__ h1out,        // [512][32][1600] (layer0 writes)
       float* __restrict__ h2out,       // [512][32][1600] (layer1 writes)
       int* __restrict__ bar)
{
  extern __shared__ char smem[];
  bf16*  sW = (bf16*)smem;                                   // [128][808]
  bf16*  sX = (bf16*)(smem + 128 * SW_STRIDE * 2);           // [128][64]
  float* sG = (float*)(smem + 128 * SW_STRIDE * 2 + 128 * KP0 * 2); // [4][32][32]
  const unsigned ldsbase = (unsigned)(size_t)smem;           // LDS byte address

  const int dir = blockIdx.x / NWG;
  const int wg  = blockIdx.x % NWG;
  const int tid  = threadIdx.x;
  const int lane = tid & 31;
  const int wv   = tid >> 5;
  const int gate = wv >> 1;
  const int mt   = wv & 1;
  const int l15  = lane & 15;
  const int kh   = lane >> 4;
  int* mybar = bar + dir * 2;

  // ---- async DMA: Whh (and Wih0) slice -> LDS, ASYNCcnt tracked ----
  {
    const bf16* wsrc = whh + (size_t)dir * (G4 * HID);
    for (int c8 = tid; c8 < 128 * 100; c8 += 256) {       // 100 b128 chunks per row
      int lc = c8 / 100, kc = (c8 - lc * 100) * 8;
      int gc = (lc >> 5) * HID + wg * 32 + (lc & 31);
      async_g2l_b128(ldsbase + (unsigned)(lc * SW_STRIDE + kc) * 2u,
                     (unsigned)(gc * HID + kc) * 2u, wsrc);
    }
    if (LAYER == 0) {
      const bf16* isrc = wih0 + (size_t)dir * (G4 * KP0);
      const unsigned xoff = (unsigned)(128 * SW_STRIDE * 2);
      for (int c8 = tid; c8 < 128 * 8; c8 += 256) {
        int lc = c8 >> 3, kc = (c8 & 7) * 8;
        int gc = (lc >> 5) * HID + wg * 32 + (lc & 31);
        async_g2l_b128(ldsbase + xoff + (unsigned)(lc * KP0 + kc) * 2u,
                       (unsigned)(gc * KP0 + kc) * 2u, isrc);
      }
    }
    wait_async0();
  }
  // ---- zero h buffer 0 (this WG's slice) ----
  {
    int b = tid >> 3, j0 = (tid & 7) * 4;
    uv2 z; z[0] = 0u; z[1] = 0u;
    *(uv2*)(hstate + ((size_t)(dir * 2 + 0) * BN + b) * HID + wg * 32 + j0) = z;
  }
  grid_barrier(mybar, NWG);   // publishes LDS (syncthreads) + h init (agent)

  float creg[4] = {0.f, 0.f, 0.f, 0.f};

  float bias0 = 0.f, bias1 = 0.f;
  if (LAYER == 0) {
    bias0 = bias[dir * G4 + gate * HID + wg * 32 + l15];
    bias1 = bias[dir * G4 + gate * HID + wg * 32 + 16 + l15];
  }

  const bf16* bcol0 = sW + (gate * 32 + l15) * SW_STRIDE;
  const bf16* bcol1 = bcol0 + 16 * SW_STRIDE;
  const bf16* gcol  = gpreT + ((size_t)dir * G4 + gate * HID + wg * 32) * (size_t)M1;

  for (int t = 0; t < LSEQ; ++t) {
    const int time = dir ? (LSEQ - 1 - t) : t;
    const int rb = t & 1, wb = rb ^ 1;

    if (LAYER == 1 && t + 1 < LSEQ) {
      // prefetch next step's C-initializer lines (global_prefetch_b8)
      int tn = dir ? (LSEQ - 2 - t) : (t + 1);
      const bf16* gpn = gcol + (size_t)l15 * M1 + (size_t)tn * BN + mt * 16 + kh * 8;
      __builtin_prefetch(gpn, 0, 1);
      __builtin_prefetch(gpn + (size_t)16 * M1, 0, 1);
    }

    v8f a0, a1;
    if (LAYER == 0) {
#pragma unroll
      for (int r = 0; r < 8; ++r) { a0[r] = bias0; a1[r] = bias1; }
    } else {
      // init C with precomputed input projection (+bias), stored transposed bf16
      const bf16* gp = gcol + (size_t)time * BN + mt * 16 + kh * 8;
      uv4 u0 = *(const uv4*)(gp + (size_t)l15 * M1);
      uv4 u1 = *(const uv4*)(gp + (size_t)(16 + l15) * M1);
      float f0[8], f1[8];
      bf8_to_f(u0, f0); bf8_to_f(u1, f1);
#pragma unroll
      for (int r = 0; r < 8; ++r) { a0[r] = f0[r]; a1[r] = f1[r]; }
    }

    // ---- recurrent GEMM: h_prev[32,800] x WhhT[800, local 32 cols], K=800 ----
    // software-pipelined: loads for kb+1 are issued before the WMMAs of kb.
    const bf16* hrow = hstate + ((size_t)(dir * 2 + rb) * BN + mt * 16 + l15) * HID;
    BfPack Ac = load_pack(hrow + kh * 16, hrow + kh * 16 + 8);
    BfPack B0c = load_pack(bcol0 + kh * 16, bcol0 + kh * 16 + 8);
    BfPack B1c = load_pack(bcol1 + kh * 16, bcol1 + kh * 16 + 8);
#pragma unroll
    for (int kb = 0; kb < 25; ++kb) {
      BfPack An, B0n, B1n;
      if (kb < 24) {
        const int ko = (kb + 1) * 32;
        An  = load_pack(hrow + ko + kh * 16, hrow + ko + kh * 16 + 8);
        B0n = load_pack(bcol0 + ko + kh * 16, bcol0 + ko + kh * 16 + 8);
        B1n = load_pack(bcol1 + ko + kh * 16, bcol1 + ko + kh * 16 + 8);
      }
      a0 = __builtin_amdgcn_wmma_f32_16x16x32_bf16(false, Ac.v, false, B0c.v, (short)0, a0, false, false);
      a1 = __builtin_amdgcn_wmma_f32_16x16x32_bf16(false, Ac.v, false, B1c.v, (short)0, a1, false, false);
      Ac = An; B0c = B0n; B1c = B1n;
    }
    if (LAYER == 0) {
      // inline x projection (K padded to 64): batch all loads, then 4 WMMAs
      const bf16* xrow = xbf + ((size_t)time * BN + mt * 16 + l15) * KP0;
      const bf16* xc0 = sX + (gate * 32 + l15) * KP0;
      const bf16* xc1 = xc0 + 16 * KP0;
      BfPack XA0 = load_pack(xrow + kh * 16, xrow + kh * 16 + 8);
      BfPack XA1 = load_pack(xrow + 32 + kh * 16, xrow + 32 + kh * 16 + 8);
      BfPack XB00 = load_pack(xc0 + kh * 16, xc0 + kh * 16 + 8);
      BfPack XB10 = load_pack(xc1 + kh * 16, xc1 + kh * 16 + 8);
      BfPack XB01 = load_pack(xc0 + 32 + kh * 16, xc0 + 32 + kh * 16 + 8);
      BfPack XB11 = load_pack(xc1 + 32 + kh * 16, xc1 + 32 + kh * 16 + 8);
      a0 = __builtin_amdgcn_wmma_f32_16x16x32_bf16(false, XA0.v, false, XB00.v, (short)0, a0, false, false);
      a1 = __builtin_amdgcn_wmma_f32_16x16x32_bf16(false, XA0.v, false, XB10.v, (short)0, a1, false, false);
      a0 = __builtin_amdgcn_wmma_f32_16x16x32_bf16(false, XA1.v, false, XB01.v, (short)0, a0, false, false);
      a1 = __builtin_amdgcn_wmma_f32_16x16x32_bf16(false, XA1.v, false, XB11.v, (short)0, a1, false, false);
    }

    // ---- exchange gates via LDS: sG[gate][batch 0..31][unit 0..31] ----
#pragma unroll
    for (int r = 0; r < 8; ++r) {
      int row = mt * 16 + kh * 8 + r;
      sG[(gate * BN + row) * 32 + l15]      = a0[r];
      sG[(gate * BN + row) * 32 + 16 + l15] = a1[r];
    }
    __syncthreads();

    // ---- LSTM cell elementwise: thread -> (batch b, 4 consecutive units) ----
    const int b = tid >> 3, j0 = (tid & 7) * 4;
    float hv[4];
#pragma unroll
    for (int q = 0; q < 4; ++q) {
      int j = j0 + q;
      float gi = sG[(0 * BN + b) * 32 + j];
      float gf = sG[(1 * BN + b) * 32 + j];
      float gg = sG[(2 * BN + b) * 32 + j];
      float go = sG[(3 * BN + b) * 32 + j];
      float c = sigm(gf) * creg[q] + sigm(gi) * tanhf(gg);
      creg[q] = c;
      hv[q] = sigm(go) * tanhf(c);
    }
    uv2 pk;
    pk[0] = bfbits(hv[0]) | (bfbits(hv[1]) << 16);
    pk[1] = bfbits(hv[2]) | (bfbits(hv[3]) << 16);
    *(uv2*)(hstate + ((size_t)(dir * 2 + wb) * BN + b) * HID + wg * 32 + j0) = pk;
    if (LAYER == 0) {
      *(uv2*)(h1out + ((size_t)time * BN + b) * (2 * HID) + dir * HID + wg * 32 + j0) = pk;
    } else {
      float4 f4; f4.x = hv[0]; f4.y = hv[1]; f4.z = hv[2]; f4.w = hv[3];
      *(float4*)(h2out + ((size_t)time * BN + b) * (2 * HID) + dir * HID + wg * 32 + j0) = f4;
    }

    grid_barrier(mybar, NWG);
  }
}

// ---------------------------------------------------------------------------
// Pre-GEMM for layer 1: G_preT[dir][3200][16384] = bias + H1[16384,1600] @ Wih1^T
// Per WG: M=128 (wave per 16 rows), N=64. The shared 64-col B slice is streamed
// through a double-buffered LDS pipeline with global_load_async_to_lds_b128:
// chunk c+1 DMAs while chunk c feeds WMMA from LDS.  Within a chunk the
// k-blocks are 1-deep software pipelined with batched B loads so each k-block
// pays one dscnt wait and issues 4 consecutive WMMAs.
// ---------------------------------------------------------------------------
__global__ void __launch_bounds__(256)
k_pregemm1(const bf16* __restrict__ h1,    // [16384][1600] bf16
           const bf16* __restrict__ wih1,  // [dir][3200][1600] bf16
           const float* __restrict__ bias, // [dir][3200]
           bf16* __restrict__ gpreT)       // [dir][3200][16384]
{
  extern __shared__ char smem[];
  bf16* sB = (bf16*)smem;                  // [2][64][160]
  const unsigned ldsB = (unsigned)(size_t)smem;

  const int bx   = blockIdx.x;
  const int dir  = bx / 6400;
  const int rr   = bx - dir * 6400;
  const int mblk = rr / 50;
  const int nblk = rr - mblk * 50;
  const int tid  = threadIdx.x;
  const int lane = tid & 31;
  const int wv   = tid >> 5;
  const int l15  = lane & 15;
  const int kh   = lane >> 4;
  const int Mbase = mblk * 128 + wv * 16;
  const int Nbase = nblk * 64;

  const bf16* wbase = wih1 + (size_t)dir * G4 * (2 * HID);

  // async-issue geometry: thread -> (col = tid&63, part = tid>>6); 5 b128 each
  const int colT  = tid & 63;
  const int partT = tid >> 6;
  const unsigned gcolB = (unsigned)((Nbase + colT) * (2 * HID)) * 2u;

  auto issue_chunk = [&](int c, int buf) {
    unsigned lds0 = ldsB + (unsigned)buf * PG_CHUNKB
                  + (unsigned)colT * (PG_CK * 2) + (unsigned)partT * 80u;
    unsigned g0 = gcolB + (unsigned)((c * PG_CK + partT * 40) * 2);
#pragma unroll
    for (int i = 0; i < 5; ++i)
      async_g2l_b128(lds0 + (unsigned)i * 16u, g0 + (unsigned)i * 16u, wbase);
  };

  v8f acc[4];
#pragma unroll
  for (int nt = 0; nt < 4; ++nt) {
    float bv = bias[dir * G4 + Nbase + nt * 16 + l15];
#pragma unroll
    for (int r = 0; r < 8; ++r) acc[nt][r] = bv;
  }

  const bf16* arow = h1 + (size_t)(Mbase + l15) * (2 * HID);

  issue_chunk(0, 0);
  for (int c = 0; c < PG_NCHUNK; ++c) {
    wait_async0();       // our chunk-c DMAs have landed
    __syncthreads();     // everyone's landed; prior buffer reads complete
    if (c + 1 < PG_NCHUNK) issue_chunk(c + 1, (c + 1) & 1);   // overlap with compute
    const bf16* bbuf = sB + (size_t)(c & 1) * (PG_COLS * PG_CK);
    const bf16* bl   = bbuf + l15 * PG_CK + kh * 16;

    // preload k-block 0 of this chunk
    BfPack Ac = load_pack(arow + c * PG_CK + kh * 16, arow + c * PG_CK + kh * 16 + 8);
    BfPack Bc[4];
#pragma unroll
    for (int nt = 0; nt < 4; ++nt)
      Bc[nt] = load_pack(bl + nt * (16 * PG_CK), bl + nt * (16 * PG_CK) + 8);

#pragma unroll
    for (int kbl = 0; kbl < 5; ++kbl) {
      BfPack An, Bn[4];
      if (kbl < 4) {
        const int kg = c * PG_CK + (kbl + 1) * 32;
        An = load_pack(arow + kg + kh * 16, arow + kg + kh * 16 + 8);
        const bf16* bn = bl + (kbl + 1) * 32;
#pragma unroll
        for (int nt = 0; nt < 4; ++nt)
          Bn[nt] = load_pack(bn + nt * (16 * PG_CK), bn + nt * (16 * PG_CK) + 8);
      }
#pragma unroll
      for (int nt = 0; nt < 4; ++nt)
        acc[nt] = __builtin_amdgcn_wmma_f32_16x16x32_bf16(false, Ac.v, false, Bc[nt].v, (short)0, acc[nt], false, false);
      Ac = An;
#pragma unroll
      for (int nt = 0; nt < 4; ++nt) Bc[nt] = Bn[nt];
    }
  }
#pragma unroll
  for (int nt = 0; nt < 4; ++nt) {
    int col = Nbase + nt * 16 + l15;
    float f[8];
#pragma unroll
    for (int r = 0; r < 8; ++r) f[r] = acc[nt][r];
    uv4 o = f_to_bf8(f);
    *(uv4*)(gpreT + ((size_t)dir * G4 + col) * (size_t)M1 + Mbase + kh * 8) = o;
  }
}

// ---------------------------------------------------------------------------
// Linear head + softmax + angle mixture -> angles[L][B][3]
// One wave per row (16384 rows). W_lin (128 KB) DMA'd to LDS asynchronously.
// ---------------------------------------------------------------------------
__global__ void __launch_bounds__(256)
k_linang(const float* __restrict__ h2, const float* __restrict__ wlin,
         const float* __restrict__ blin, const float* __restrict__ alpha,
         float* __restrict__ angles)
{
  extern __shared__ float sWl[];  // [20][1600]
  const int tid = threadIdx.x;
  {
    const unsigned ldsW = (unsigned)(size_t)sWl;
    for (int i = tid; i < 8000; i += 256)   // 8000 x b128 = 32000 floats
      async_g2l_b128(ldsW + (unsigned)i * 16u, (unsigned)i * 16u, wlin);
    wait_async0();
  }
  __syncthreads();

  const int lane = tid & 31;
  const int wv   = tid >> 5;
  const int row  = blockIdx.x * 8 + wv;

  const float* hr = h2 + (size_t)row * 1600;
  float accv[20];
#pragma unroll
  for (int a = 0; a < 20; ++a) accv[a] = 0.f;
  for (int kk = 0; kk < 50; ++kk) {
    float hv = hr[kk * 32 + lane];
#pragma unroll
    for (int a = 0; a < 20; ++a) accv[a] += hv * sWl[a * 1600 + kk * 32 + lane];
  }
  float mylog = 0.f;
#pragma unroll
  for (int a = 0; a < 20; ++a) {
    float v = accv[a];
    v += __shfl_xor(v, 16); v += __shfl_xor(v, 8); v += __shfl_xor(v, 4);
    v += __shfl_xor(v, 2);  v += __shfl_xor(v, 1);
    if (lane == a) mylog = v + blin[a];
  }
  float mx = -1e30f;
  for (int a = 0; a < 20; ++a) mx = fmaxf(mx, __shfl(mylog, a));
  float e = (lane < 20) ? expf(mylog - mx) : 0.f;
  float ssum = e;
  ssum += __shfl_xor(ssum, 16); ssum += __shfl_xor(ssum, 8); ssum += __shfl_xor(ssum, 4);
  ssum += __shfl_xor(ssum, 2);  ssum += __shfl_xor(ssum, 1);
  float p = e / ssum;

  float sv[3] = {0.f, 0.f, 0.f}, cv[3] = {0.f, 0.f, 0.f};
  if (lane < 20) {
#pragma unroll
    for (int c = 0; c < 3; ++c) {
      float al = alpha[lane * 3 + c];
      sv[c] = p * sinf(al);
      cv[c] = p * cosf(al);
    }
  }
#pragma unroll
  for (int c = 0; c < 3; ++c) {
    float s = sv[c], q = cv[c];
    s += __shfl_xor(s, 16); s += __shfl_xor(s, 8); s += __shfl_xor(s, 4);
    s += __shfl_xor(s, 2);  s += __shfl_xor(s, 1);
    q += __shfl_xor(q, 16); q += __shfl_xor(q, 8); q += __shfl_xor(q, 4);
    q += __shfl_xor(q, 2);  q += __shfl_xor(q, 1);
    sv[c] = s; cv[c] = q;
  }
  if (lane < 3) angles[(size_t)row * 3 + lane] = atan2f(sv[lane], cv[lane]);
}

// ---------------------------------------------------------------------------
// NeRF chain extension: 32 independent batch chains, 1536 sequential steps.
// ---------------------------------------------------------------------------
__global__ void k_nerf(const float* __restrict__ angles, float* __restrict__ out)
{
  const int b = threadIdx.x;  // 0..31
  const float PI = 3.14159265358979323846f;
  const float BL[3] = {1.458f, 1.525f, 1.329f};
  const float BA[3] = {2.124f, 1.941f, 2.028f};
  float cth[3], sth[3];
#pragma unroll
  for (int i = 0; i < 3; ++i) { float th = PI - BA[i]; cth[i] = cosf(th); sth[i] = sinf(th); }

  float Ax = 0.f, Ay = 0.f, Az = 0.f;
  float Bx = BL[0], By = 0.f, Bz = 0.f;
  float th0 = PI - BA[0];
  float Cx = Bx + BL[1] * cosf(th0), Cy = BL[1] * sinf(th0), Cz = 0.f;

  for (int n = 0; n < 3 * LSEQ; ++n) {
    int l = n / 3, a3 = n - 3 * l;
    float phi = angles[((size_t)l * BN + b) * 3 + a3];
    float vx = Cx - Bx, vy = Cy - By, vz = Cz - Bz;
    float inv = 1.f / (sqrtf(vx * vx + vy * vy + vz * vz) + 1e-8f);
    float bcx = vx * inv, bcy = vy * inv, bcz = vz * inv;
    float ux = Bx - Ax, uy = By - Ay, uz = Bz - Az;
    float nx = uy * bcz - uz * bcy, ny = uz * bcx - ux * bcz, nz = ux * bcy - uy * bcx;
    float ninv = 1.f / (sqrtf(nx * nx + ny * ny + nz * nz) + 1e-8f);
    nx *= ninv; ny *= ninv; nz *= ninv;
    float mx = ny * bcz - nz * bcy, my = nz * bcx - nx * bcz, mz = nx * bcy - ny * bcx;
    float cp = cosf(phi), sp = sinf(phi);
    float d = BL[a3], ct = cth[a3], st = sth[a3];
    float nxp = Cx + d * (ct * bcx + st * (cp * mx + sp * nx));
    float nyp = Cy + d * (ct * bcy + st * (cp * my + sp * ny));
    float nzp = Cz + d * (ct * bcz + st * (cp * mz + sp * nz));
    size_t o = ((size_t)n * BN + b) * 3;
    out[o + 0] = nxp; out[o + 1] = nyp; out[o + 2] = nzp;
    Ax = Bx; Ay = By; Az = Bz;
    Bx = Cx; By = Cy; Bz = Cz;
    Cx = nxp; Cy = nyp; Cz = nzp;
  }
}

// ---------------------------------------------------------------------------
// host side
// ---------------------------------------------------------------------------
static inline size_t ws_alloc(size_t& off, size_t bytes) {
  size_t o = off;
  off = (off + bytes + 255) & ~(size_t)255;
  return o;
}

extern "C" void kernel_launch(void* const* d_in, const int* in_sizes, int n_in,
                              void* d_out, int out_size, void* d_ws, size_t ws_size,
                              hipStream_t stream) {
  (void)in_sizes; (void)n_in; (void)out_size; (void)ws_size;

  const float* x     = (const float*)d_in[0];
  const float* Wih[4]  = {(const float*)d_in[1], (const float*)d_in[5],
                          (const float*)d_in[9], (const float*)d_in[13]};
  const float* Whh[4]  = {(const float*)d_in[2], (const float*)d_in[6],
                          (const float*)d_in[10], (const float*)d_in[14]};
  const float* bih[4]  = {(const float*)d_in[3], (const float*)d_in[7],
                          (const float*)d_in[11], (const float*)d_in[15]};
  const float* bhh[4]  = {(const float*)d_in[4], (const float*)d_in[8],
                          (const float*)d_in[12], (const float*)d_in[16]};
  const float* Wlin  = (const float*)d_in[17];
  const float* blin  = (const float*)d_in[18];
  const float* alpha = (const float*)d_in[19];
  float* out = (float*)d_out;
  char* ws = (char*)d_ws;

  size_t off = 0;
  const size_t whh_elems  = (size_t)G4 * HID;          // per matrix
  const size_t wih0_elems = (size_t)G4 * KP0;
  const size_t wih1_elems = (size_t)G4 * 2 * HID;
  size_t o_whh  = ws_alloc(off, 4 * whh_elems * 2);    // [l0f,l0b,l1f,l1b] bf16
  size_t o_wih0 = ws_alloc(off, 2 * wih0_elems * 2);   // [f,b] bf16 (padded K)
  size_t o_wih1 = ws_alloc(off, 2 * wih1_elems * 2);   // [f,b] bf16
  size_t o_xbf  = ws_alloc(off, (size_t)M1 * KP0 * 2); // padded x bf16
  size_t o_bias = ws_alloc(off, 4 * (size_t)G4 * 4);   // combined biases f32
  size_t o_hst  = ws_alloc(off, 4 * (size_t)BN * HID * 2);      // h double buffers
  size_t o_h1   = ws_alloc(off, (size_t)M1 * 2 * HID * 2);      // H1 bf16
  size_t o_h2   = ws_alloc(off, (size_t)M1 * 2 * HID * 4);      // H2 f32
  size_t o_gpre = ws_alloc(off, 2 * (size_t)G4 * M1 * 2);       // G_preT bf16
  size_t o_ang  = ws_alloc(off, (size_t)M1 * 3 * 4);            // angles f32
  size_t o_bar  = ws_alloc(off, 64);

  bf16* whh_bf  = (bf16*)(ws + o_whh);
  bf16* wih0_bf = (bf16*)(ws + o_wih0);
  bf16* wih1_bf = (bf16*)(ws + o_wih1);
  bf16* xbf     = (bf16*)(ws + o_xbf);
  float* bias   = (float*)(ws + o_bias);
  bf16* hst     = (bf16*)(ws + o_hst);
  bf16* h1      = (bf16*)(ws + o_h1);
  float* h2     = (float*)(ws + o_h2);
  bf16* gpreT   = (bf16*)(ws + o_gpre);
  float* ang    = (float*)(ws + o_ang);
  int* bar      = (int*)(ws + o_bar);

  auto cvt = [&](const float* src, bf16* dst, int rows, int scols, int dcols) {
    int total = rows * dcols;
    k_cvt_pad<<<(total + 255) / 256, 256, 0, stream>>>(src, dst, rows, scols, dcols);
  };

  // weights -> bf16 (Wih0 / x zero-padded K 42 -> 64)
  for (int i = 0; i < 4; ++i)
    cvt(Whh[i], whh_bf + i * whh_elems, G4, HID, HID);
  cvt(Wih[0], wih0_bf + 0 * wih0_elems, G4, 42, KP0);
  cvt(Wih[1], wih0_bf + 1 * wih0_elems, G4, 42, KP0);
  cvt(Wih[2], wih1_bf + 0 * wih1_elems, G4, 2 * HID, 2 * HID);
  cvt(Wih[3], wih1_bf + 1 * wih1_elems, G4, 2 * HID, 2 * HID);
  cvt(x, xbf, M1, 42, KP0);
  for (int i = 0; i < 4; ++i)
    k_bias_sum<<<(G4 + 255) / 256, 256, 0, stream>>>(bih[i], bhh[i], bias + i * G4, G4);
  k_init_bar<<<1, 64, 0, stream>>>(bar);

  const size_t lstm_lds = 128 * SW_STRIDE * 2 + 128 * KP0 * 2 + 4 * BN * 32 * 4;
  const size_t pg_lds   = 2 * PG_CHUNKB;

  // layer 0 (bias at [0..1], x-proj inline)
  k_lstm<0><<<2 * NWG, 256, lstm_lds, stream>>>(
      whh_bf, wih0_bf, xbf, gpreT, bias, hst, h1, h2, bar);

  // layer 1 input projection (big WMMA GEMM, async-LDS pipelined), bias folded in
  k_pregemm1<<<2 * 128 * 50, 256, pg_lds, stream>>>(h1, wih1_bf, bias + 2 * G4, gpreT);

  // layer 1 recurrence
  k_lstm<1><<<2 * NWG, 256, lstm_lds, stream>>>(
      whh_bf + 2 * whh_elems, wih0_bf, xbf, gpreT, bias + 2 * G4, hst, h1, h2, bar);

  // head: linear + softmax + angle mixture
  k_linang<<<M1 / 8, 256, 20 * 1600 * sizeof(float), stream>>>(h2, Wlin, blin, alpha, ang);

  // NeRF chain extension
  k_nerf<<<1, 32, 0, stream>>>(ang, out);
}
